// BaseLSTM_8881992368553
// MI455X (gfx1250) — matrix operand order repaired and checked by reading, not compiled, not run on hardware
//
#include <hip/hip_runtime.h>
#include <hip/hip_bf16.h>
#include <stdint.h>
#include <stddef.h>

#define T_STEPS 1024
#define BATCH   128
#define HID     512
#define GATES   2048            // 4*HID
#define KTOT    1024            // I + H
#define ASTR    520             // LDS row stride in bf16 elems (512 + 8 pad -> 4-dword bank skew)
#define NBLK    32
#define INV_KEEP 1.4285714285714286f

typedef __attribute__((ext_vector_type(16))) __bf16 v16bf;
typedef __attribute__((ext_vector_type(8)))  __bf16 v8bf;
typedef __attribute__((ext_vector_type(8)))  float  v8f;
typedef __attribute__((ext_vector_type(4)))  unsigned int u32x4;
typedef __attribute__((ext_vector_type(8)))  int    i32x8;
typedef __attribute__((ext_vector_type(4)))  int    i32x4;

#if defined(__gfx1250__) && __has_builtin(__builtin_amdgcn_tensor_load_to_lds)
#define USE_TDM 1
#else
#define USE_TDM 0
#endif

struct SMem {
  __bf16 Ax[16 * ASTR];   // K = 0..511 half of A tile (x_t or h1_new)
  __bf16 Ah[16 * ASTR];   // K = 512..1023 half (h_prev)
};

__device__ __forceinline__ float sigf(float x) { return 1.0f / (1.0f + __expf(-x)); }

// ---------------- device-wide sense barrier (persistent grid) ----------------
__device__ __forceinline__ void grid_sync(int* arrive, int* gen, int target) {
  __threadfence();
  __syncthreads();
  if (threadIdx.x == 0) {
    if (atomicAdd(arrive, 1) == NBLK - 1) {
      atomicExch(arrive, 0);
      __threadfence();
      atomicAdd(gen, 1);
    } else {
      while (__hip_atomic_load(gen, __ATOMIC_ACQUIRE, __HIP_MEMORY_SCOPE_AGENT) < target)
        __builtin_amdgcn_s_sleep(2);
    }
  }
  __syncthreads();
}

// ---------------- LDS staging ----------------
// 16 rows x 512 cols fp32 -> bf16 into Ax (padded rows)
__device__ __forceinline__ void stage_x_f32(SMem& sm, const float* __restrict__ xrow) {
  for (int c = threadIdx.x; c < 16 * 64; c += 256) {
    int r = c >> 6, col = (c & 63) * 8;
    const float4* s = (const float4*)(xrow + r * HID + col);
    float4 f0 = s[0], f1 = s[1];
    v8bf v;
    v[0] = (__bf16)f0.x; v[1] = (__bf16)f0.y; v[2] = (__bf16)f0.z; v[3] = (__bf16)f0.w;
    v[4] = (__bf16)f1.x; v[5] = (__bf16)f1.y; v[6] = (__bf16)f1.z; v[7] = (__bf16)f1.w;
    *(v8bf*)(&sm.Ax[r * ASTR + col]) = v;
  }
}

// 16 rows x 512 cols bf16 global -> LDS (padded rows), manual path
__device__ __forceinline__ void stage_h_copy(__bf16* dst, const __bf16* __restrict__ src) {
  for (int c = threadIdx.x; c < 16 * 64; c += 256) {
    int r = c >> 6, col = (c & 63) * 8;
    *(v8bf*)(dst + r * ASTR + col) = *(const v8bf*)(src + r * HID + col);
  }
}

#if USE_TDM
// TDM: DMA 16x512 bf16 tile from global into LDS, padding 4 dwords after every
// 256 dwords (= one 512-elem bf16 row) to reproduce the ASTR=520 row stride.
__device__ __forceinline__ void stage_h_tdm(__bf16* dstLDS, const __bf16* srcGlobal,
                                            const SMem& smbase) {
  if (threadIdx.x == 0) {
    uint32_t ldsOff = (uint32_t)((const char*)dstLDS - (const char*)&smbase);
    uint64_t ga = (uint64_t)(uintptr_t)srcGlobal;
    u32x4 g0;
    g0[0] = 1u;                                            // count=1, user desc
    g0[1] = ldsOff;                                        // lds_addr
    g0[2] = (uint32_t)ga;                                  // global_addr[31:0]
    g0[3] = (uint32_t)((ga >> 32) & 0x01FFFFFFu) | (2u << 30);  // addr[56:32] | type=2
    i32x8 g1;
    g1[0] = (1 << 16)       // data_size = 2 bytes
          | (1 << 20)       // pad_enable
          | (7 << 22)       // pad_interval: 256 dwords
          | (3 << 25);      // pad_amount: 4 dwords (8 bf16)
    g1[1] = (int)(512u << 16);   // tensor_dim0 = 512
    g1[2] = (int)(128u << 16);   // tensor_dim1 = 128
    g1[3] = (int)(512u << 16);   // tile_dim0 = 512
    g1[4] = 16;                  // tile_dim1 = 16
    g1[5] = 512;                 // tensor_dim0_stride = 512
    g1[6] = 0; g1[7] = 0;
    i32x4 g2 = {0, 0, 0, 0}, g3 = {0, 0, 0, 0};
    i32x8 g4 = {0, 0, 0, 0, 0, 0, 0, 0};   // extra group (6-arg clang-23 form)
    __builtin_amdgcn_tensor_load_to_lds(g0, g1, g2, g3, g4, 0);
    __builtin_amdgcn_s_wait_tensorcnt(0);
  }
}
#endif

// ---------------- fused gate-GEMM + LSTM cell for one 16x16 (b,h) tile ----------------
__device__ __forceinline__ void gemm_cell(const SMem& sm,
                                          const __bf16* __restrict__ WbT,  // [KTOT][GATES] bf16
                                          float bI, float bF, float bG, float bO,
                                          const float* drp, float* cloc,
                                          int lane, int m0, int nh0,
                                          __bf16* __restrict__ Hnext,
                                          float* __restrict__ outp) {
  const int lm = lane & 15;
  const int lsel = lane >> 4;
  v8f ai = {}, af = {}, ag = {}, ao = {};
#pragma unroll 4
  for (int kc = 0; kc < KTOT / 32; ++kc) {
    const int kk = kc * 32;
    const __bf16* Abase = (kk < HID) ? &sm.Ax[0] : &sm.Ah[0];
    const int ko = (kk < HID) ? kk : (kk - HID);
    // A fragment: lane (lsel half) holds K = {ko..ko+7} and {ko+16..ko+23} (+8 for upper half)
    v8bf alo = *(const v8bf*)(Abase + lm * ASTR + ko + lsel * 8);
    v8bf ahi = *(const v8bf*)(Abase + lm * ASTR + ko + 16 + lsel * 8);
    v16bf a = __builtin_shufflevector(alo, ahi, 0, 1, 2, 3, 4, 5, 6, 7,
                                      8, 9, 10, 11, 12, 13, 14, 15);
    // B fragments: lane = N column (nh0 + lm); K in same order as A fragment
    const __bf16* brow = WbT + (size_t)(kk + lane) * GATES;
    v16bf b0, b1, b2, b3;
#pragma unroll
    for (int s = 0; s < 16; ++s) {
      const int k = kk + ((s < 8) ? (lsel * 8 + s) : (16 + lsel * 8 + (s - 8)));
      const __bf16* bp = WbT + (size_t)k * GATES + nh0 + lm;
      b0[s] = bp[0 * HID];
      b1[s] = bp[1 * HID];
      b2[s] = bp[2 * HID];
      b3[s] = bp[3 * HID];
    }
    __builtin_prefetch(brow + (size_t)32 * GATES, 0, 1);
    ai = __builtin_amdgcn_wmma_f32_16x16x32_bf16(false, a, false, b0, (short)0, ai, false, false);
    af = __builtin_amdgcn_wmma_f32_16x16x32_bf16(false, a, false, b1, (short)0, af, false, false);
    ag = __builtin_amdgcn_wmma_f32_16x16x32_bf16(false, a, false, b2, (short)0, ag, false, false);
    ao = __builtin_amdgcn_wmma_f32_16x16x32_bf16(false, a, false, b3, (short)0, ao, false, false);
  }
  // cell update: C/D layout -> element r: b = m0 + r + 8*lsel, h = nh0 + lm
#pragma unroll
  for (int r = 0; r < 8; ++r) {
    float gi = ai[r] + bI, gf = af[r] + bF, gg = ag[r] + bG, go = ao[r] + bO;
    float c = sigf(gf) * cloc[r] + sigf(gi) * tanhf(gg);
    float h = sigf(go) * tanhf(c);
    float d = drp[r];
    c *= d; h *= d;
    cloc[r] = c;
    const int b = m0 + r + 8 * lsel;
    const int hc = nh0 + lm;
    Hnext[b * HID + hc] = (__bf16)h;
    if (outp) outp[(size_t)b * HID + hc] = h;
  }
}

// ---------------- persistent 2-layer LSTM ----------------
__global__ __launch_bounds__(256, 1)
void lstm_persistent(const float* __restrict__ xs,
                     const __bf16* __restrict__ WbT1, const __bf16* __restrict__ WbT2,
                     const float* __restrict__ bias1, const float* __restrict__ bias2,
                     const float* __restrict__ dropbuf,
                     __bf16* __restrict__ Hb1,   // [2][BATCH*HID]
                     __bf16* __restrict__ Hb2,   // [2][BATCH*HID]
                     float* __restrict__ out,    // [T][BATCH][HID]
                     int* __restrict__ bar) {
  __shared__ SMem sm;
  const int lane = threadIdx.x & 31;
  const int wave = threadIdx.x >> 5;
  const int mi = blockIdx.x >> 2;                    // 0..7 batch tile
  const int m0 = mi * 16;
  const int nh0 = (((blockIdx.x & 3) * 8) + wave) * 16;  // 0..496 hidden tile
  const int lm = lane & 15, lsel = lane >> 4;

  // time-invariant per-lane constants
  const float b1I = bias1[0 * HID + nh0 + lm], b1F = bias1[1 * HID + nh0 + lm];
  const float b1G = bias1[2 * HID + nh0 + lm], b1O = bias1[3 * HID + nh0 + lm];
  const float b2I = bias2[0 * HID + nh0 + lm], b2F = bias2[1 * HID + nh0 + lm];
  const float b2G = bias2[2 * HID + nh0 + lm], b2O = bias2[3 * HID + nh0 + lm];
  float drp[8], c1loc[8], c2loc[8];
#pragma unroll
  for (int r = 0; r < 8; ++r) {
    drp[r] = dropbuf[(m0 + r + 8 * lsel) * HID + nh0 + lm];
    c1loc[r] = 0.0f;
    c2loc[r] = 0.0f;
  }

  int p = 0, target = 0;
#pragma unroll 1
  for (int t = 0; t < T_STEPS; ++t) {
    // ---- layer 1: A = [x_t | h1_prev] ----
    stage_x_f32(sm, xs + (size_t)t * BATCH * HID + (size_t)m0 * HID);
#if USE_TDM
    stage_h_tdm(sm.Ah, Hb1 + (size_t)p * BATCH * HID + (size_t)m0 * HID, sm);
#else
    stage_h_copy(sm.Ah, Hb1 + (size_t)p * BATCH * HID + (size_t)m0 * HID);
#endif
    __syncthreads();
    gemm_cell(sm, WbT1, b1I, b1F, b1G, b1O, drp, c1loc, lane, m0, nh0,
              Hb1 + (size_t)(1 - p) * BATCH * HID, nullptr);
    grid_sync(bar, bar + 1, ++target);

    // ---- layer 2: A = [h1_new | h2_prev] ----
    stage_h_copy(sm.Ax, Hb1 + (size_t)(1 - p) * BATCH * HID + (size_t)m0 * HID);
#if USE_TDM
    stage_h_tdm(sm.Ah, Hb2 + (size_t)p * BATCH * HID + (size_t)m0 * HID, sm);
#else
    stage_h_copy(sm.Ah, Hb2 + (size_t)p * BATCH * HID + (size_t)m0 * HID);
#endif
    __syncthreads();
    gemm_cell(sm, WbT2, b2I, b2F, b2G, b2O, drp, c2loc, lane, m0, nh0,
              Hb2 + (size_t)(1 - p) * BATCH * HID,
              out + (size_t)t * BATCH * HID);
    grid_sync(bar, bar + 1, ++target);
    p ^= 1;
  }
}

// ---------------- prep kernels ----------------
// WbT[k][n] = (k<512 ? W_ih[n][k] : W_hh[n][k-512]) as bf16  (K-major for WMMA B)
__global__ void k_prep_w(const float* __restrict__ Wih, const float* __restrict__ Whh,
                         __bf16* __restrict__ WbT) {
  int idx = blockIdx.x * blockDim.x + threadIdx.x;
  if (idx >= KTOT * GATES) return;
  int k = idx / GATES, n = idx % GATES;
  float v = (k < HID) ? Wih[(size_t)n * HID + k] : Whh[(size_t)n * HID + (k - HID)];
  WbT[idx] = (__bf16)v;
}

__global__ void k_prep_misc(const float* __restrict__ bih1, const float* __restrict__ bhh1,
                            const float* __restrict__ bih2, const float* __restrict__ bhh2,
                            const float* __restrict__ mask,
                            float* __restrict__ bias1, float* __restrict__ bias2,
                            float* __restrict__ dropbuf,
                            __bf16* __restrict__ Hb1, __bf16* __restrict__ Hb2,
                            int* __restrict__ bar) {
  int idx = blockIdx.x * blockDim.x + threadIdx.x;
  if (idx < GATES) {
    bias1[idx] = bih1[idx] + bhh1[idx];
    bias2[idx] = bih2[idx] + bhh2[idx];
  }
  if (idx < BATCH * HID) {
    dropbuf[idx] = mask[idx] * INV_KEEP;
    Hb1[idx] = (__bf16)0.0f;   // zero ping buffer (p=0)
    Hb2[idx] = (__bf16)0.0f;
  }
  if (idx == 0) { bar[0] = 0; bar[1] = 0; }
}

// ---------------- host entry ----------------
extern "C" void kernel_launch(void* const* d_in, const int* in_sizes, int n_in,
                              void* d_out, int out_size, void* d_ws, size_t ws_size,
                              hipStream_t stream) {
  const float* xs    = (const float*)d_in[0];
  const float* W_ih1 = (const float*)d_in[1];
  const float* W_hh1 = (const float*)d_in[2];
  const float* b_ih1 = (const float*)d_in[3];
  const float* b_hh1 = (const float*)d_in[4];
  const float* W_ih2 = (const float*)d_in[5];
  const float* W_hh2 = (const float*)d_in[6];
  const float* b_ih2 = (const float*)d_in[7];
  const float* b_hh2 = (const float*)d_in[8];
  const float* mask  = (const float*)d_in[9];
  float* out = (float*)d_out;

  char* w = (char*)d_ws;
  __bf16* WbT1 = (__bf16*)w;               w += (size_t)KTOT * GATES * 2;    // 4 MB
  __bf16* WbT2 = (__bf16*)w;               w += (size_t)KTOT * GATES * 2;    // 4 MB
  __bf16* Hb1  = (__bf16*)w;               w += (size_t)2 * BATCH * HID * 2; // 256 KB
  __bf16* Hb2  = (__bf16*)w;               w += (size_t)2 * BATCH * HID * 2; // 256 KB
  float*  bias1 = (float*)w;               w += (size_t)GATES * 4;
  float*  bias2 = (float*)w;               w += (size_t)GATES * 4;
  float*  dropbuf = (float*)w;             w += (size_t)BATCH * HID * 4;     // 256 KB
  int*    bar = (int*)w;

  const int wElems = KTOT * GATES;
  k_prep_w<<<(wElems + 255) / 256, 256, 0, stream>>>(W_ih1, W_hh1, WbT1);
  k_prep_w<<<(wElems + 255) / 256, 256, 0, stream>>>(W_ih2, W_hh2, WbT2);
  k_prep_misc<<<(BATCH * HID + 255) / 256, 256, 0, stream>>>(
      b_ih1, b_hh1, b_ih2, b_hh2, mask, bias1, bias2, dropbuf, Hb1, Hb2, bar);
  lstm_persistent<<<NBLK, 256, 0, stream>>>(xs, WbT1, WbT2, bias1, bias2, dropbuf,
                                            Hb1, Hb2, out, bar);
}
